// HeteroStateModelEncoder_43542378447338
// MI455X (gfx1250) — matrix operand order repaired and checked by reading, not compiled, not run on hardware
//
#include <hip/hip_runtime.h>
#include <hip/hip_bf16.h>
#include <math.h>

// ---------------------------------------------------------------------------
// MI455X (gfx1250) implementation of the hetero-GNN reference.
//  - All dense transforms via v_wmma_f32_16x16x32_f16 (f16 in, f32 acc).
//  - X tile + W staged in LDS as f16 (W transposed) -> branch-free inner loop:
//    wide ds_load fragments + WMMA only.
//  - Epilogue: LDS round-trip -> coalesced float4 global stores.
//  - Edge propagation: per-(edge,feature) coalesced scatter + f32 atomics.
// ---------------------------------------------------------------------------

typedef __attribute__((ext_vector_type(16))) _Float16 v16h;
typedef __attribute__((ext_vector_type(8)))  _Float16 v8h;
typedef __attribute__((ext_vector_type(8)))  float    v8f;

// ---------------------------- utility kernels ------------------------------

__global__ void fill0_kernel(float* __restrict__ p, long long n) {
  long long i = (long long)blockIdx.x * blockDim.x + threadIdx.x;
  if (i < n) p[i] = 0.f;
}

__global__ void count_deg_kernel(const int* __restrict__ dst, int E, float* __restrict__ deg) {
  int i = blockIdx.x * blockDim.x + threadIdx.x;
  if (i < E) atomicAdd(&deg[dst[i]], 1.0f);
}

__global__ void count_deg_rel_kernel(const int* __restrict__ dst, const int* __restrict__ et,
                                     int E, float* __restrict__ degrel, int N) {
  int i = blockIdx.x * blockDim.x + threadIdx.x;
  if (i < E) atomicAdd(&degrel[(long long)et[i] * N + dst[i]], 1.0f);
}

__global__ void rsqrt_inplace_kernel(float* __restrict__ deg, int N) {
  int i = blockIdx.x * blockDim.x + threadIdx.x;
  if (i < N) { float d = deg[i]; deg[i] = d > 0.f ? rsqrtf(d) : 0.f; }
}

__global__ void edge_norm_kernel(const int* __restrict__ src, const int* __restrict__ dst,
                                 const float* __restrict__ dinv, float* __restrict__ norm, int E) {
  int i = blockIdx.x * blockDim.x + threadIdx.x;
  if (i < E) norm[i] = dinv[src[i]] * dinv[dst[i]];
}

// Xout[dst[e], f] += Xin[src[e], f] * norm[e]
__global__ void spmm_scatter_kernel(const int* __restrict__ src, const int* __restrict__ dst,
                                    const float* __restrict__ norm,
                                    const float* __restrict__ Xin, float* __restrict__ Xout,
                                    int E, int F) {
  long long i = (long long)blockIdx.x * blockDim.x + threadIdx.x;
  if (i < (long long)E * F) {
    int e = (int)(i / F);
    int f = (int)(i - (long long)e * F);
    atomicAdd(&Xout[(long long)dst[e] * F + f], Xin[(long long)src[e] * F + f] * norm[e]);
  }
}

// Xout[dst[e], f] += Xin[src[e], f] / max(deg[dst[e]], 1)   (segment mean)
__global__ void scatter_mean_kernel(const int* __restrict__ src, const int* __restrict__ dst,
                                    const float* __restrict__ deg,
                                    const float* __restrict__ Xin, float* __restrict__ Xout,
                                    int E, int F) {
  long long i = (long long)blockIdx.x * blockDim.x + threadIdx.x;
  if (i < (long long)E * F) {
    int e = (int)(i / F);
    int f = (int)(i - (long long)e * F);
    int d = dst[e];
    float w = 1.0f / fmaxf(deg[d], 1.0f);
    atomicAdd(&Xout[(long long)d * F + f], Xin[(long long)src[e] * F + f] * w);
  }
}

// RGCN relation-r mean scatter of pre-transformed features
__global__ void rgcn_scatter_kernel(const int* __restrict__ src, const int* __restrict__ dst,
                                    const int* __restrict__ et, int r,
                                    const float* __restrict__ degr,
                                    const float* __restrict__ Xin, float* __restrict__ Xout, int E) {
  long long i = (long long)blockIdx.x * blockDim.x + threadIdx.x;
  if (i < (long long)E * 64) {
    int e = (int)(i >> 6);
    int f = (int)(i & 63);
    if (et[e] == r) {
      int d = dst[e];
      float w = 1.0f / fmaxf(degr[d], 1.0f);
      atomicAdd(&Xout[(long long)d * 64 + f], Xin[(long long)src[e] * 64 + f] * w);
    }
  }
}

// ResGatedGraphConv edge kernel: gate = sigmoid(k[dst]+q[src]+ea@We); out += gate*v[src]
__global__ void resgate_scatter_kernel(const int* __restrict__ src, const int* __restrict__ dst,
                                       const float* __restrict__ ea,
                                       const float* __restrict__ Kb, const float* __restrict__ Qb,
                                       const float* __restrict__ Vb, const float* __restrict__ We,
                                       float* __restrict__ Out, int E) {
  long long i = (long long)blockIdx.x * blockDim.x + threadIdx.x;
  if (i < (long long)E * 64) {
    int e = (int)(i >> 6);
    int h = (int)(i & 63);
    int s = src[e], d = dst[e];
    float eh = ea[2 * e] * We[h] + ea[2 * e + 1] * We[64 + h];
    float z = Kb[(long long)d * 64 + h] + Qb[(long long)s * 64 + h] + eh;
    float g = 1.0f / (1.0f + __expf(-z));
    atomicAdd(&Out[(long long)d * 64 + h], g * Vb[(long long)s * 64 + h]);
  }
}

__global__ void l2norm_rows_kernel(float* __restrict__ X, int N, int F) {
  int r = blockIdx.x * blockDim.x + threadIdx.x;
  if (r < N) {
    float s = 0.f;
    for (int j = 0; j < F; ++j) { float v = X[(long long)r * F + j]; s += v * v; }
    float inv = 1.0f / fmaxf(sqrtf(s), 1e-12f);
    for (int j = 0; j < F; ++j) X[(long long)r * F + j] *= inv;
  }
}

// ------------------------------ WMMA GEMM ----------------------------------
// Out[N,FOUT] = X[N,FIN] @ W[FIN,FOUT] (+bias if !ACCUM) (+=Out if ACCUM) (ReLU)
// Block = 256 threads = 8 waves; each wave owns a 16-row x FOUT strip.
// X tile (128 x FINP) and W (transposed, FOUT x FINP) staged in LDS as f16
// with K padded to FINP (32 or 64): branch-free K-loop of wide ds_loads +
// v_wmma_f32_16x16x32_f16. Rows address-clamped (not zeroed): D row m depends
// only on A row m, so clamped garbage rows never reach a stored output.
// Epilogue: accumulators -> wave-private LDS tile -> coalesced float4 stores.
template <int FIN, int FOUT, bool ACCUM, bool RELU>
__global__ __launch_bounds__(256) void wmma_gemm_kernel(
    const float* __restrict__ X, const float* __restrict__ W,
    const float* __restrict__ bias, float* __restrict__ Out, int N) {
  constexpr int FINP = (FIN <= 32) ? 32 : 64;  // K padded to WMMA depth
  constexpr int NT = FOUT / 16;
  constexpr int ROWS = 128;  // 8 waves * 16 rows

  __shared__ _Float16 WS[FOUT * FINP];   // transposed: WS[c*FINP + k]
  __shared__ _Float16 XS[ROWS * FINP];   // XS[row*FINP + k]
  __shared__ float    OUTS[ROWS * FOUT]; // wave-private epilogue staging

  const int rows0 = blockIdx.x * ROWS;

  // ---- stage W (transposed, K zero-padded); unconditional clamped loads ----
#pragma unroll 4
  for (int idx = threadIdx.x; idx < FOUT * FINP; idx += 256) {
    int c = idx / FINP;
    int k = idx & (FINP - 1);
    int kk = (k < FIN) ? k : (FIN - 1);
    float v = W[kk * FOUT + c];
    if (FIN != FINP) v = (k < FIN) ? v : 0.f;  // select, not branch
    WS[c * FINP + k] = (_Float16)v;
  }

  // ---- stage X tile (row-clamped, K zero-padded) ----
  if constexpr (FIN == FINP) {
#pragma unroll 4
    for (int idx = threadIdx.x; idx < ROWS * FINP / 4; idx += 256) {
      int row = idx / (FINP / 4);
      int k4 = (idx - row * (FINP / 4)) * 4;
      int grow = rows0 + row;
      int gr = (grow < N) ? grow : (N - 1);
      const float4 v = *(const float4*)(X + (long long)gr * FIN + k4);
      _Float16* d = &XS[row * FINP + k4];
      d[0] = (_Float16)v.x; d[1] = (_Float16)v.y;
      d[2] = (_Float16)v.z; d[3] = (_Float16)v.w;
    }
  } else {
#pragma unroll 4
    for (int idx = threadIdx.x; idx < ROWS * FINP; idx += 256) {
      int row = idx / FINP;
      int k = idx & (FINP - 1);
      int grow = rows0 + row;
      int gr = (grow < N) ? grow : (N - 1);
      int kk = (k < FIN) ? k : 0;
      float v = X[(long long)gr * FIN + kk];
      v = (k < FIN) ? v : 0.f;  // select, not branch
      XS[row * FINP + k] = (_Float16)v;
    }
  }
  __syncthreads();

  const int lane = threadIdx.x & 31;
  const int warp = threadIdx.x >> 5;
  const int row0 = rows0 + warp * 16;
  if (row0 >= N) return;  // uniform per wave; EXEC full at every WMMA

  const int r    = lane & 15;
  const int hsel = lane >> 4;
  const int kha  = hsel * 8;   // unused: A also loads via khb (lanes 16-31 hold K+16..31)
  const int khb  = hsel * 16;  // B: lanes 16-31 hold K+16..31 rows
  const int col  = lane & 15;

  v8f zero = {};
  v8f acc[NT];
#pragma unroll
  for (int t = 0; t < NT; ++t) acc[t] = zero;

#pragma unroll
  for (int kc = 0; kc < FINP; kc += 32) {
    const _Float16* xrow = &XS[(warp * 16 + r) * FINP + kc];
    v16h a = *(const v16h*)(xrow + khb);
#pragma unroll
    for (int t = 0; t < NT; ++t) {
      v16h b = *(const v16h*)&WS[(t * 16 + col) * FINP + kc + khb];
      acc[t] = __builtin_amdgcn_wmma_f32_16x16x32_f16(
          false, a, false, b, (short)0, acc[t], false, false);
    }
  }

  // ---- epilogue: dump C fragments to wave-private LDS tile ----
  float* outs = &OUTS[warp * 16 * FOUT];
  const int mbase = hsel * 8;  // C/D layout: vgpr v -> row mbase+v
#pragma unroll
  for (int t = 0; t < NT; ++t) {
#pragma unroll
    for (int v = 0; v < 8; ++v) {
      outs[(mbase + v) * FOUT + t * 16 + col] = acc[t][v];
    }
  }

  // ---- coalesced float4 writeback (bias / accumulate / relu fused) ----
#pragma unroll
  for (int i = lane; i < 16 * FOUT / 4; i += 32) {
    int rl = i / (FOUT / 4);
    int c4 = (i - rl * (FOUT / 4)) * 4;
    int m = row0 + rl;
    if (m < N) {
      float v0 = outs[rl * FOUT + c4 + 0];
      float v1 = outs[rl * FOUT + c4 + 1];
      float v2 = outs[rl * FOUT + c4 + 2];
      float v3 = outs[rl * FOUT + c4 + 3];
      if constexpr (!ACCUM) {  // bias fused on the non-accumulating pass
        const float4 bv = *(const float4*)(bias + c4);
        v0 += bv.x; v1 += bv.y; v2 += bv.z; v3 += bv.w;
      }
      float* op = Out + (long long)m * FOUT + c4;
      if constexpr (ACCUM) {
        const float4 ov = *(const float4*)op;
        v0 += ov.x; v1 += ov.y; v2 += ov.z; v3 += ov.w;
      }
      if constexpr (RELU) {
        v0 = fmaxf(v0, 0.f); v1 = fmaxf(v1, 0.f);
        v2 = fmaxf(v2, 0.f); v3 = fmaxf(v3, 0.f);
      }
      float4 res; res.x = v0; res.y = v1; res.z = v2; res.w = v3;
      *(float4*)op = res;
    }
  }
}

// ------------------------------- head --------------------------------------

__global__ void head_logits_kernel(const float* __restrict__ H, const float* __restrict__ lw,
                                   const float* __restrict__ lb, float* __restrict__ y, int N) {
  int r = blockIdx.x * blockDim.x + threadIdx.x;
  if (r < N) {
    float acc = lb[0];
#pragma unroll
    for (int j = 0; j < 32; ++j) acc += H[(long long)r * 32 + j] * lw[j];
    y[r] = acc;
  }
}

__global__ void reduce_max_lse_kernel(const float* __restrict__ y, int N, float* __restrict__ red) {
  __shared__ float sm[1024];
  int t = threadIdx.x;
  float m = -3.4e38f;
  for (int i = t; i < N; i += blockDim.x) m = fmaxf(m, y[i]);
  sm[t] = m; __syncthreads();
  for (int s = 512; s > 0; s >>= 1) { if (t < s) sm[t] = fmaxf(sm[t], sm[t + s]); __syncthreads(); }
  float mx = sm[0]; __syncthreads();
  float a = 0.f;
  for (int i = t; i < N; i += blockDim.x) a += __expf(y[i] - mx);
  sm[t] = a; __syncthreads();
  for (int s = 512; s > 0; s >>= 1) { if (t < s) sm[t] += sm[t + s]; __syncthreads(); }
  if (t == 0) { red[0] = mx; red[1] = logf(sm[0]); }
}

__global__ void logsoftmax_kernel(const float* __restrict__ y, const float* __restrict__ red,
                                  float* __restrict__ out, int N) {
  int r = blockIdx.x * blockDim.x + threadIdx.x;
  if (r < N) out[r] = y[r] - red[0] - red[1];
}

// ------------------------------ launcher -----------------------------------

static inline unsigned gcnt(long long n, int b) { return (unsigned)((n + b - 1) / b); }

static void gemm64(const float* X, const float* W, const float* b, float* O,
                   int N, int Fin, bool accum, hipStream_t st) {
  dim3 grid(gcnt(N, 128)), blk(256);
  if (Fin == 7) {
    if (accum) wmma_gemm_kernel<7, 64, true,  false><<<grid, blk, 0, st>>>(X, W, b, O, N);
    else       wmma_gemm_kernel<7, 64, false, false><<<grid, blk, 0, st>>>(X, W, b, O, N);
  } else {
    if (accum) wmma_gemm_kernel<64, 64, true,  false><<<grid, blk, 0, st>>>(X, W, b, O, N);
    else       wmma_gemm_kernel<64, 64, false, false><<<grid, blk, 0, st>>>(X, W, b, O, N);
  }
}

extern "C" void kernel_launch(void* const* d_in, const int* in_sizes, int n_in,
                              void* d_out, int out_size, void* d_ws, size_t ws_size,
                              hipStream_t stream) {
  (void)n_in; (void)out_size; (void)ws_size;
  // Input order: setup_inputs() dict order; nested 'params' flattened with
  // JAX's sorted-key tree order (uppercase sorts before lowercase).
  const float* x_game   = (const float*)d_in[0];
  const float* x_state  = (const float*)d_in[1];
  const float* ea_hist  = (const float*)d_in[2];
  const float* last_W   = (const float*)d_in[3];
  const float* last_b   = (const float*)d_in[4];
  const float* lin_W    = (const float*)d_in[5];
  const float* lin_b    = (const float*)d_in[6];
  const float* rg_We    = (const float*)d_in[7];
  const float* rg_Wk    = (const float*)d_in[8];
  const float* rg_Wq    = (const float*)d_in[9];
  const float* rg_Wskip = (const float*)d_in[10];
  const float* rg_Wv    = (const float*)d_in[11];
  const float* rg_b     = (const float*)d_in[12];
  const float* rg_bk    = (const float*)d_in[13];
  const float* rg_bq    = (const float*)d_in[14];
  const float* rg_bv    = (const float*)d_in[15];
  const float* rgcn_Wrel  = (const float*)d_in[16];
  const float* rgcn_Wroot = (const float*)d_in[17];
  const float* rgcn_b     = (const float*)d_in[18];
  const float* s3_Wl = (const float*)d_in[19];
  const float* s3_Wr = (const float*)d_in[20];
  const float* s3_bl = (const float*)d_in[21];
  const float* s4_Wl = (const float*)d_in[22];
  const float* s4_Wr = (const float*)d_in[23];
  const float* s4_bl = (const float*)d_in[24];
  const float* s5_Wl = (const float*)d_in[25];
  const float* s5_Wr = (const float*)d_in[26];
  const float* s5_bl = (const float*)d_in[27];
  const float* s7_Wl = (const float*)d_in[28];
  const float* s7_Wr = (const float*)d_in[29];
  const float* s7_bl = (const float*)d_in[30];
  const float* tag1_W = (const float*)d_in[31];
  const float* tag1_b = (const float*)d_in[32];
  const float* tag6_W = (const float*)d_in[33];
  const float* tag6_b = (const float*)d_in[34];
  const int* gg_src  = (const int*)d_in[35];
  const int* gg_dst  = (const int*)d_in[36];
  const int* gg_type = (const int*)d_in[37];
  const int* hist_src = (const int*)d_in[38];
  const int* hist_dst = (const int*)d_in[39];
  const int* in_src   = (const int*)d_in[40];
  const int* in_dst   = (const int*)d_in[41];
  const int* par_src  = (const int*)d_in[42];
  const int* par_dst  = (const int*)d_in[43];

  const int NG   = in_sizes[0] / 7;
  const int NS   = in_sizes[1] / 7;
  const int EH   = in_sizes[2] / 2;
  const int EGG  = in_sizes[35];
  const int EIN  = in_sizes[40];
  const int EPAR = in_sizes[42];

  // bump allocator over d_ws
  char* base = (char*)d_ws;
  size_t off = 0;
  auto alloc = [&](long long nf) -> float* {
    float* p = (float*)(base + off);
    off += (size_t)((nf * 4 + 255) & ~255LL);
    return p;
  };
  float* deg_g   = alloc(NG);            // gg degree -> dinv (in place)
  float* norm_gg = alloc(EGG);
  float* h7a     = alloc((long long)NG * 7);
  float* h7b     = alloc((long long)NG * 7);
  float* g1      = alloc((long long)NG * 64);  // tag1 output
  float* degrel  = alloc((long long)3 * NG);
  float* tmp64   = alloc((long long)NG * 64);  // g1 @ Wrel[r]
  float* g2      = alloc((long long)NG * 64);  // rgcn output (final game emb)
  float* kb      = alloc((long long)NS * 64);
  float* qb      = alloc((long long)NG * 64);
  float* vb      = alloc((long long)NG * 64);
  float* s1      = alloc((long long)NS * 64);  // res_gated out
  float* deg_h   = alloc(NS);
  float* aggH    = alloc((long long)NS * 64);
  float* s2      = alloc((long long)NS * 64);
  float* deg_in  = alloc(NS);
  float* aggI    = alloc((long long)NS * 64);
  float* s3b     = alloc((long long)NS * 64);
  float* s4b     = alloc((long long)NS * 64);
  float* deg_p   = alloc(NS);                  // -> dinv for tag6 norm
  float* degc_p  = alloc(NS);                  // raw counts for sage7 mean
  float* norm_par = alloc(EPAR);
  float* hpa     = alloc((long long)NS * 64);
  float* hpb     = alloc((long long)NS * 64);
  float* s5buf   = alloc((long long)NS * 64);  // tag6 out
  float* aggP    = alloc((long long)NS * 64);
  float* s6b     = alloc((long long)NS * 64);  // sage7 out
  float* hbuf    = alloc((long long)NS * 32);
  float* ylog    = alloc(NS);
  float* red     = alloc(2);

  auto zero = [&](float* p, long long n) {
    fill0_kernel<<<gcnt(n, 256), 256, 0, stream>>>(p, n);
  };

  // ---- Stage A: TAGConv(K=3) on game graph (7-wide propagation) ----
  zero(deg_g, NG);
  count_deg_kernel<<<gcnt(EGG, 256), 256, 0, stream>>>(gg_dst, EGG, deg_g);
  rsqrt_inplace_kernel<<<gcnt(NG, 256), 256, 0, stream>>>(deg_g, NG);
  edge_norm_kernel<<<gcnt(EGG, 256), 256, 0, stream>>>(gg_src, gg_dst, deg_g, norm_gg, EGG);
  gemm64(x_game, tag1_W + 0 * 7 * 64, tag1_b, g1, NG, 7, false, stream);
  zero(h7b, (long long)NG * 7);
  spmm_scatter_kernel<<<gcnt((long long)EGG * 7, 256), 256, 0, stream>>>(gg_src, gg_dst, norm_gg, x_game, h7b, EGG, 7);
  gemm64(h7b, tag1_W + 1 * 7 * 64, nullptr, g1, NG, 7, true, stream);
  zero(h7a, (long long)NG * 7);
  spmm_scatter_kernel<<<gcnt((long long)EGG * 7, 256), 256, 0, stream>>>(gg_src, gg_dst, norm_gg, h7b, h7a, EGG, 7);
  gemm64(h7a, tag1_W + 2 * 7 * 64, nullptr, g1, NG, 7, true, stream);
  zero(h7b, (long long)NG * 7);
  spmm_scatter_kernel<<<gcnt((long long)EGG * 7, 256), 256, 0, stream>>>(gg_src, gg_dst, norm_gg, h7a, h7b, EGG, 7);
  gemm64(h7b, tag1_W + 3 * 7 * 64, nullptr, g1, NG, 7, true, stream);

  // ---- Stage B: RGCN (3 relations, mean agg) on game graph ----
  zero(degrel, (long long)3 * NG);
  count_deg_rel_kernel<<<gcnt(EGG, 256), 256, 0, stream>>>(gg_dst, gg_type, EGG, degrel, NG);
  gemm64(g1, rgcn_Wroot, rgcn_b, g2, NG, 64, false, stream);
  for (int r = 0; r < 3; ++r) {
    gemm64(g1, rgcn_Wrel + (long long)r * 64 * 64, nullptr, tmp64, NG, 64, false, stream);
    rgcn_scatter_kernel<<<gcnt((long long)EGG * 64, 256), 256, 0, stream>>>(
        gg_src, gg_dst, gg_type, r, degrel + (long long)r * NG, tmp64, g2, EGG);
  }

  // ---- Stage C: ResGated game->state over hist edges ----
  gemm64(x_state, rg_Wk, rg_bk, kb, NS, 7, false, stream);
  gemm64(g2, rg_Wq, rg_bq, qb, NG, 64, false, stream);
  gemm64(g2, rg_Wv, rg_bv, vb, NG, 64, false, stream);
  gemm64(x_state, rg_Wskip, rg_b, s1, NS, 7, false, stream);  // skip + bias first
  resgate_scatter_kernel<<<gcnt((long long)EH * 64, 256), 256, 0, stream>>>(
      hist_src, hist_dst, ea_hist, kb, qb, vb, rg_We, s1, EH);

  // ---- Stage D: SAGE3 over hist edges (src = g2) ----
  zero(deg_h, NS);
  count_deg_kernel<<<gcnt(EH, 256), 256, 0, stream>>>(hist_dst, EH, deg_h);
  zero(aggH, (long long)NS * 64);
  scatter_mean_kernel<<<gcnt((long long)EH * 64, 256), 256, 0, stream>>>(hist_src, hist_dst, deg_h, g2, aggH, EH, 64);
  gemm64(aggH, s3_Wl, s3_bl, s2, NS, 64, false, stream);
  gemm64(s1, s3_Wr, nullptr, s2, NS, 64, true, stream);
  l2norm_rows_kernel<<<gcnt(NS, 256), 256, 0, stream>>>(s2, NS, 64);

  // ---- Stage E: SAGE4 + SAGE5 over 'in' edges (shared agg of g2) ----
  zero(deg_in, NS);
  count_deg_kernel<<<gcnt(EIN, 256), 256, 0, stream>>>(in_dst, EIN, deg_in);
  zero(aggI, (long long)NS * 64);
  scatter_mean_kernel<<<gcnt((long long)EIN * 64, 256), 256, 0, stream>>>(in_src, in_dst, deg_in, g2, aggI, EIN, 64);
  gemm64(aggI, s4_Wl, s4_bl, s3b, NS, 64, false, stream);
  gemm64(s2, s4_Wr, nullptr, s3b, NS, 64, true, stream);
  l2norm_rows_kernel<<<gcnt(NS, 256), 256, 0, stream>>>(s3b, NS, 64);
  gemm64(aggI, s5_Wl, s5_bl, s4b, NS, 64, false, stream);
  gemm64(s3b, s5_Wr, nullptr, s4b, NS, 64, true, stream);
  l2norm_rows_kernel<<<gcnt(NS, 256), 256, 0, stream>>>(s4b, NS, 64);

  // ---- Stage F: TAGConv(K=2) on par edges (64-wide propagation) ----
  zero(deg_p, NS);
  count_deg_kernel<<<gcnt(EPAR, 256), 256, 0, stream>>>(par_dst, EPAR, deg_p);
  rsqrt_inplace_kernel<<<gcnt(NS, 256), 256, 0, stream>>>(deg_p, NS);
  edge_norm_kernel<<<gcnt(EPAR, 256), 256, 0, stream>>>(par_src, par_dst, deg_p, norm_par, EPAR);
  gemm64(s4b, tag6_W + 0 * 64 * 64, tag6_b, s5buf, NS, 64, false, stream);
  zero(hpa, (long long)NS * 64);
  spmm_scatter_kernel<<<gcnt((long long)EPAR * 64, 256), 256, 0, stream>>>(par_src, par_dst, norm_par, s4b, hpa, EPAR, 64);
  gemm64(hpa, tag6_W + 1 * 64 * 64, nullptr, s5buf, NS, 64, true, stream);
  zero(hpb, (long long)NS * 64);
  spmm_scatter_kernel<<<gcnt((long long)EPAR * 64, 256), 256, 0, stream>>>(par_src, par_dst, norm_par, hpa, hpb, EPAR, 64);
  gemm64(hpb, tag6_W + 2 * 64 * 64, nullptr, s5buf, NS, 64, true, stream);

  // ---- Stage G: SAGE7 over par edges (src = dst = s5buf) ----
  zero(degc_p, NS);
  count_deg_kernel<<<gcnt(EPAR, 256), 256, 0, stream>>>(par_dst, EPAR, degc_p);
  zero(aggP, (long long)NS * 64);
  scatter_mean_kernel<<<gcnt((long long)EPAR * 64, 256), 256, 0, stream>>>(par_src, par_dst, degc_p, s5buf, aggP, EPAR, 64);
  gemm64(aggP, s7_Wl, s7_bl, s6b, NS, 64, false, stream);
  gemm64(s5buf, s7_Wr, nullptr, s6b, NS, 64, true, stream);
  l2norm_rows_kernel<<<gcnt(NS, 256), 256, 0, stream>>>(s6b, NS, 64);

  // ---- Stage H: head 64->32 (ReLU, WMMA) -> 32->1 -> log_softmax(axis=0) ----
  wmma_gemm_kernel<64, 32, false, true><<<gcnt(NS, 128), 256, 0, stream>>>(s6b, lin_W, lin_b, hbuf, NS);
  head_logits_kernel<<<gcnt(NS, 256), 256, 0, stream>>>(hbuf, last_W, last_b, ylog, NS);
  reduce_max_lse_kernel<<<1, 1024, 0, stream>>>(ylog, NS, red);
  logsoftmax_kernel<<<gcnt(NS, 256), 256, 0, stream>>>(ylog, red, (float*)d_out, NS);
}
